// TriangleMultiplicationOutgoing_61400852464436
// MI455X (gfx1250) — matrix operand order repaired and checked by reading, not compiled, not run on hardware
//
#include <hip/hip_runtime.h>

// ---------------------------------------------------------------------------
// TriangleMultiplicationOutgoing for MI455X (gfx1250), wave32 + WMMA bf16.
// B=1, N=384, C=M=128. All GEMMs via v_wmma_f32_16x16x32_bf16.
// Operand layouts used here (A and B share one contiguous K order):
//   A 16x32 bf16 : lane L (r=L&15, h=L>>4) holds row r, K = h*16..h*16+15
//                  -> one contiguous 32B load per lane.
//   B 32x16 bf16 : lane L holds column (L&15), K = (L>>4)*16 .. +15
//                  -> one contiguous 32B load per lane (B given row-major
//                     over [col][k], i.e. our transposed-weight / right_t).
//   C/D 16x16 f32: VGPR g of lane L = (row g + 8*(L>>4), col L&15).
// ---------------------------------------------------------------------------

typedef __attribute__((ext_vector_type(16))) __bf16 v16bf;
typedef __attribute__((ext_vector_type(8)))  __bf16 v8bf;
typedef __attribute__((ext_vector_type(4)))  __bf16 v4bf;
typedef __attribute__((ext_vector_type(8)))  float  v8f;
typedef __attribute__((ext_vector_type(4)))  float  f32x4;

#define NDIM 384
#define CDIM 128
#define NN   147456   // N*N

static __device__ __forceinline__ __bf16 f2bf(float f) {
    unsigned int u = __builtin_bit_cast(unsigned int, f);
    u += 0x7FFFu + ((u >> 16) & 1u);                  // round-to-nearest-even
    unsigned short s = (unsigned short)(u >> 16);
    return __builtin_bit_cast(__bf16, s);
}
static __device__ __forceinline__ float bf2f(__bf16 h) {
    unsigned int u = ((unsigned int)__builtin_bit_cast(unsigned short, h)) << 16;
    return __builtin_bit_cast(float, u);
}
static __device__ __forceinline__ float sigmoidf(float x) {
    return 1.0f / (1.0f + __expf(-x));
}

static __device__ __forceinline__ v8f wmma_bf16(v16bf a, v16bf b, v8f c) {
    return __builtin_amdgcn_wmma_f32_16x16x32_bf16(
        /*neg_a=*/false, a, /*neg_b=*/false, b,
        /*c_mod=*/(short)0, c, /*reuse_a=*/false, /*reuse_b=*/false);
}

// A tile 16x32: base points at row0 of a row-major [*, stride] bf16 matrix.
static __device__ __forceinline__ v16bf load_a_tile(const __bf16* base, int stride,
                                                    int kk, int lane) {
    const __bf16* p = base + (size_t)(lane & 15) * stride + kk + ((lane >> 4) * 16);
    return *(const v16bf*)p;
}

// B tile 32x16: colbase is row-major over [col][k]; col = col0 + (lane&15).
static __device__ __forceinline__ v16bf load_b_tile(const __bf16* colbase, int stride,
                                                    int col0, int kk, int lane) {
    const __bf16* p = colbase + (size_t)(col0 + (lane & 15)) * stride
                              + kk + ((lane >> 4) * 16);
    return *(const v16bf*)p;
}

// ---------------------------------------------------------------------------
// Kernel 0: transpose + bf16-convert the six 128x128 weight matrices.
// Output order in wt: [Wl_t, Wgl_t, Wr_t, Wgr_t, Wg_t, Wo_t], each [out][in].
// ---------------------------------------------------------------------------
__global__ void prep_weights_kernel(const float* __restrict__ Wl,
                                    const float* __restrict__ Wr,
                                    const float* __restrict__ Wgl,
                                    const float* __restrict__ Wgr,
                                    const float* __restrict__ Wg,
                                    const float* __restrict__ Wo,
                                    __bf16* __restrict__ wt) {
    int idx = blockIdx.x * blockDim.x + threadIdx.x;        // 6*16384 threads
    int widx = idx >> 14;
    int o = idx & 16383;
    int m = o >> 7, c = o & 127;
    const float* src;
    switch (widx) {
        case 0: src = Wl;  break;
        case 1: src = Wgl; break;
        case 2: src = Wr;  break;
        case 3: src = Wgr; break;
        case 4: src = Wg;  break;
        default: src = Wo; break;
    }
    wt[(size_t)widx * 16384 + m * 128 + c] = f2bf(src[c * 128 + m]);
}

// ---------------------------------------------------------------------------
// Kernel 1/4a: LayerNorm over 128 channels, f32 in -> bf16 out.
// One wave per row (4 channels per lane), 8 rows per 256-thread block.
// ---------------------------------------------------------------------------
__global__ void ln_bf16_kernel(const float* __restrict__ x,
                               const float* __restrict__ g,
                               const float* __restrict__ b,
                               __bf16* __restrict__ y, int rows) {
    int r = blockIdx.x * 8 + (threadIdx.x >> 5);
    if (r >= rows) return;
    int lane = threadIdx.x & 31;
    int c0 = lane * 4;
    f32x4 xv = *(const f32x4*)(x + (size_t)r * CDIM + c0);
    float s = xv[0] + xv[1] + xv[2] + xv[3];
    #pragma unroll
    for (int off = 16; off > 0; off >>= 1) s += __shfl_xor(s, off, 32);
    float mu = s * (1.0f / 128.0f);
    float d0 = xv[0]-mu, d1 = xv[1]-mu, d2 = xv[2]-mu, d3 = xv[3]-mu;
    float q = d0*d0 + d1*d1 + d2*d2 + d3*d3;
    #pragma unroll
    for (int off = 16; off > 0; off >>= 1) q += __shfl_xor(q, off, 32);
    float inv = rsqrtf(q * (1.0f / 128.0f) + 1e-5f);
    v4bf o;
    o[0] = f2bf(d0 * inv * g[c0+0] + b[c0+0]);
    o[1] = f2bf(d1 * inv * g[c0+1] + b[c0+1]);
    o[2] = f2bf(d2 * inv * g[c0+2] + b[c0+2]);
    o[3] = f2bf(d3 * inv * g[c0+3] + b[c0+3]);
    *(v4bf*)(y + (size_t)r * CDIM + c0) = o;
}

// ---------------------------------------------------------------------------
// Kernel 2: projections. blockIdx.y = mode (0=left, 1=right, 2=gate).
// Block = 256 threads = 8 waves covering a 64-row x 128-col tile:
//   wave w: rows r0 = blk*64 + (w>>1)*16, cols (w&1)*64 .. +63 (4 WMMA tiles).
// mode 0/1: dual GEMM (linear + sigmoid gate), epilogue mask*lin*sig,
//           stored channel-major (dst[m*NN + row]) with one 16B store/lane.
// mode 2:   single GEMM, sigmoid, stored row-major bf16 gate[row*C + m].
// ---------------------------------------------------------------------------
__global__ void proj_kernel(const __bf16* __restrict__ xln,
                            const float*  __restrict__ mask,
                            const __bf16* __restrict__ wt,
                            const float* __restrict__ bl, const float* __restrict__ bgl,
                            const float* __restrict__ br, const float* __restrict__ bgr,
                            const float* __restrict__ bg,
                            __bf16* __restrict__ left_t,
                            __bf16* __restrict__ right_t,
                            __bf16* __restrict__ gate) {
    const int lane = threadIdx.x & 31;
    const int w = threadIdx.x >> 5;
    const int mode = blockIdx.y;
    const int r0 = blockIdx.x * 64 + (w >> 1) * 16;
    const int m0 = (w & 1) * 64;
    const __bf16* A = xln + (size_t)r0 * CDIM;
    const int half = lane >> 4, nc = lane & 15;

    if (mode == 2) {
        const __bf16* W = wt + 4 * 16384;                 // Wg_t
        v8f acc[4] = {};
        for (int kk = 0; kk < CDIM; kk += 32) {
            v16bf a = load_a_tile(A, CDIM, kk, lane);
            #pragma unroll
            for (int t = 0; t < 4; ++t) {
                v16bf bm = load_b_tile(W, CDIM, m0 + t * 16, kk, lane);
                acc[t] = wmma_bf16(a, bm, acc[t]);
            }
        }
        #pragma unroll
        for (int t = 0; t < 4; ++t) {
            int m = m0 + t * 16 + nc;
            float bb = bg[m];
            #pragma unroll
            for (int v = 0; v < 8; ++v) {
                int row = r0 + 8 * half + v;
                gate[(size_t)row * CDIM + m] = f2bf(sigmoidf(acc[t][v] + bb));
            }
        }
    } else {
        const __bf16* W0 = wt + (size_t)(mode == 0 ? 0 : 2) * 16384;   // Wl_t / Wr_t
        const __bf16* W1 = W0 + 16384;                                 // Wgl_t / Wgr_t
        const float* b0 = (mode == 0) ? bl  : br;
        const float* b1 = (mode == 0) ? bgl : bgr;
        __bf16* dst = (mode == 0) ? left_t : right_t;
        v8f accL[4] = {}; v8f accS[4] = {};
        for (int kk = 0; kk < CDIM; kk += 32) {
            v16bf a = load_a_tile(A, CDIM, kk, lane);
            #pragma unroll
            for (int t = 0; t < 4; ++t) {
                v16bf bLm = load_b_tile(W0, CDIM, m0 + t * 16, kk, lane);
                accL[t] = wmma_bf16(a, bLm, accL[t]);
                v16bf bSm = load_b_tile(W1, CDIM, m0 + t * 16, kk, lane);
                accS[t] = wmma_bf16(a, bSm, accS[t]);
            }
        }
        #pragma unroll
        for (int t = 0; t < 4; ++t) {
            int m = m0 + t * 16 + nc;
            float bb0 = b0[m], bb1 = b1[m];
            v8bf pack;
            #pragma unroll
            for (int v = 0; v < 8; ++v) {
                int row = r0 + 8 * half + v;
                float lin = accL[t][v] + bb0;
                float sg  = sigmoidf(accS[t][v] + bb1);
                pack[v] = f2bf(mask[row] * lin * sg);
            }
            // channel-major store: dst[m][row], rows contiguous along v
            *(v8bf*)(dst + (size_t)m * NN + r0 + 8 * half) = pack;
        }
    }
}

// ---------------------------------------------------------------------------
// Kernel 3: triangle contraction. For channel c: Out_c = L_c @ R_c^T.
// left_t/right_t are [c][row][k] bf16 (row length N=384), so A rows and the
// K x 16 B operand (columns j of R_c) are both contiguous 32B lane loads.
// Each wave computes a 32x32 output tile; 8 waves/block; 18 blocks/channel.
// ---------------------------------------------------------------------------
__global__ void tri_kernel(const __bf16* __restrict__ left_t,
                           const __bf16* __restrict__ right_t,
                           float* __restrict__ out_pre) {
    const int lane = threadIdx.x & 31;
    const int w = threadIdx.x >> 5;
    const int c  = blockIdx.x / 18;
    const int bi = blockIdx.x % 18;
    const int tile = bi * 8 + w;          // 0..143
    const int ti = tile / 12, tj = tile % 12;
    const int i0 = ti * 32, j0 = tj * 32;
    const __bf16* Lc = left_t  + (size_t)c * NN;
    const __bf16* Rc = right_t + (size_t)c * NN;

    v8f acc00 = {}, acc01 = {}, acc10 = {}, acc11 = {};
    for (int kk = 0; kk < NDIM; kk += 32) {
        v16bf a0 = load_a_tile(Lc + (size_t)i0 * NDIM,        NDIM, kk, lane);
        v16bf a1 = load_a_tile(Lc + (size_t)(i0 + 16) * NDIM, NDIM, kk, lane);
        v16bf b0 = load_b_tile(Rc, NDIM, j0,      kk, lane);
        v16bf b1 = load_b_tile(Rc, NDIM, j0 + 16, kk, lane);
        acc00 = wmma_bf16(a0, b0, acc00);
        acc01 = wmma_bf16(a0, b1, acc01);
        acc10 = wmma_bf16(a1, b0, acc10);
        acc11 = wmma_bf16(a1, b1, acc11);
    }
    const int half = lane >> 4, nj = lane & 15;
    #pragma unroll
    for (int ih = 0; ih < 2; ++ih) {
        #pragma unroll
        for (int jh = 0; jh < 2; ++jh) {
            v8f acc = ih ? (jh ? acc11 : acc10) : (jh ? acc01 : acc00);
            int j = j0 + jh * 16 + nj;
            #pragma unroll
            for (int v = 0; v < 8; ++v) {
                int i = i0 + ih * 16 + 8 * half + v;
                out_pre[((size_t)i * NDIM + j) * CDIM + c] = acc[v];
            }
        }
    }
}

// ---------------------------------------------------------------------------
// Kernel 5: final projection: out = (LN2 @ Wo + bo) * gate, f32 output.
// Same tiling as proj_kernel mode 2.
// ---------------------------------------------------------------------------
__global__ void out_kernel(const __bf16* __restrict__ ln2,
                           const __bf16* __restrict__ wo_t,
                           const float* __restrict__ bo,
                           const __bf16* __restrict__ gate,
                           float* __restrict__ out) {
    const int lane = threadIdx.x & 31;
    const int w = threadIdx.x >> 5;
    const int r0 = blockIdx.x * 64 + (w >> 1) * 16;
    const int m0 = (w & 1) * 64;
    const __bf16* A = ln2 + (size_t)r0 * CDIM;

    v8f acc[4] = {};
    for (int kk = 0; kk < CDIM; kk += 32) {
        v16bf a = load_a_tile(A, CDIM, kk, lane);
        #pragma unroll
        for (int t = 0; t < 4; ++t) {
            v16bf bm = load_b_tile(wo_t, CDIM, m0 + t * 16, kk, lane);
            acc[t] = wmma_bf16(a, bm, acc[t]);
        }
    }
    const int half = lane >> 4, nc = lane & 15;
    #pragma unroll
    for (int t = 0; t < 4; ++t) {
        int m = m0 + t * 16 + nc;
        float bb = bo[m];
        #pragma unroll
        for (int v = 0; v < 8; ++v) {
            int row = r0 + 8 * half + v;
            float gv = bf2f(gate[(size_t)row * CDIM + m]);
            out[(size_t)row * CDIM + m] = (acc[t][v] + bb) * gv;
        }
    }
}

// ---------------------------------------------------------------------------
// Host launcher. Workspace layout (all offsets 256B-aligned):
//   wt      :  6 * 128*128 bf16   =    196,608 B
//   xln/ln2 :  NN*128 bf16        = 37,748,736 B   (reused for LN2 output)
//   left_t  :  NN*128 bf16        = 37,748,736 B
//   right_t :  NN*128 bf16        = 37,748,736 B
//   gate    :  NN*128 bf16        = 37,748,736 B
//   out_pre :  NN*128 f32         = 75,497,472 B
//   total   ~ 226.7 MB
// ---------------------------------------------------------------------------
extern "C" void kernel_launch(void* const* d_in, const int* in_sizes, int n_in,
                              void* d_out, int out_size, void* d_ws, size_t ws_size,
                              hipStream_t stream) {
    const float* act  = (const float*)d_in[0];
    const float* mask = (const float*)d_in[1];
    const float* ln_g = (const float*)d_in[2];
    const float* ln_b = (const float*)d_in[3];
    const float* Wl   = (const float*)d_in[4];
    const float* bl   = (const float*)d_in[5];
    const float* Wr   = (const float*)d_in[6];
    const float* br   = (const float*)d_in[7];
    const float* Wgl  = (const float*)d_in[8];
    const float* bgl  = (const float*)d_in[9];
    const float* Wgr  = (const float*)d_in[10];
    const float* bgr  = (const float*)d_in[11];
    const float* cg   = (const float*)d_in[12];
    const float* cb   = (const float*)d_in[13];
    const float* Wo   = (const float*)d_in[14];
    const float* bo   = (const float*)d_in[15];
    const float* Wg   = (const float*)d_in[16];
    const float* bg   = (const float*)d_in[17];
    float* out = (float*)d_out;

    char* ws = (char*)d_ws;
    const size_t SZ_BF = (size_t)NN * CDIM * 2;
    __bf16* wt      = (__bf16*)(ws);
    __bf16* xln     = (__bf16*)(ws + 196608);
    __bf16* left_t  = (__bf16*)(ws + 196608 + SZ_BF);
    __bf16* right_t = (__bf16*)(ws + 196608 + 2 * SZ_BF);
    __bf16* gate    = (__bf16*)(ws + 196608 + 3 * SZ_BF);
    float*  out_pre = (float*) (ws + 196608 + 4 * SZ_BF);

    // 0) transpose weights to bf16 [out][in]
    prep_weights_kernel<<<(6 * 16384) / 256, 256, 0, stream>>>(Wl, Wr, Wgl, Wgr, Wg, Wo, wt);
    // 1) LayerNorm(act) -> bf16
    ln_bf16_kernel<<<NN / 8, 256, 0, stream>>>(act, ln_g, ln_b, xln, NN);
    // 2) left / right / gate projections (WMMA)
    proj_kernel<<<dim3(NN / 64, 3), 256, 0, stream>>>(xln, mask, wt, bl, bgl, br, bgr, bg,
                                                      left_t, right_t, gate);
    // 3) triangle contraction: 128 channels x 18 blocks (WMMA)
    tri_kernel<<<128 * 18, 256, 0, stream>>>(left_t, right_t, out_pre);
    // 4) LayerNorm(out_pre) -> bf16 (reuse xln buffer)
    ln_bf16_kernel<<<NN / 8, 256, 0, stream>>>(out_pre, cg, cb, xln, NN);
    // 5) output projection * gate -> f32 (WMMA)
    out_kernel<<<NN / 64, 256, 0, stream>>>(xln, wt + 5 * 16384, bo, gate, out);
}
